// Decoder_53403623358494
// MI455X (gfx1250) — compile-verified, matrix operand order checked by reading, not run on hardware
//
#include <hip/hip_runtime.h>
#include <hip/hip_bf16.h>
#include <stdint.h>
#include <stddef.h>

// ---------------- types ----------------
typedef __bf16 bf16;
typedef __attribute__((ext_vector_type(16))) __bf16 bfx16;
typedef __attribute__((ext_vector_type(8)))  __bf16 bfx8;
typedef __attribute__((ext_vector_type(8)))  float  fx8;

union BF16x16 { bfx16 v; bfx8 h[2]; };

// ---------------- model dims ----------------
static constexpr int Bb   = 32;
static constexpr int TENC = 400;
static constexpr int TMEL = 800;
static constexpr int NMEL = 80;
static constexpr int EDIM = 768;
static constexpr int PRE  = 256;
static constexpr int RNN  = 1024;
static constexpr int GATES= 4096;
static constexpr int ADIM = 128;
static constexpr int LF   = 32;
static constexpr int LK   = 31;

static constexpr int KATT = PRE + EDIM + RNN;     // 2048 : [pre | ctx | ah]
static constexpr int KDEC = RNN + EDIM + RNN;     // 2816 : [ah | ctx | dh]

static constexpr int NWG   = 128;    // persistent grid (assumed co-resident)
static constexpr int TPB   = 256;    // 8 wave32 per WG
static constexpr int NWAVE = 8;
static constexpr int KS_ATT = KATT / 32 / NWAVE;  // 8  k-steps per wave
static constexpr int KS_DEC = KDEC / 32 / NWAVE;  // 11 k-steps per wave

// ---------------- workspace layout (bytes, 256-aligned) ----------------
static constexpr size_t OFF_WATT  = 256;
static constexpr size_t SZ_WATT   = (size_t)GATES * KATT * 2;         // 16.8 MB bf16
static constexpr size_t OFF_WDEC  = OFF_WATT + SZ_WATT;
static constexpr size_t SZ_WDEC   = (size_t)GATES * KDEC * 2;         // 23.1 MB bf16
static constexpr size_t OFF_MEMB  = OFF_WDEC + SZ_WDEC;
static constexpr size_t SZ_MEMB   = (size_t)Bb * TENC * EDIM * 2;     // 19.7 MB bf16
static constexpr size_t OFF_XS    = OFF_MEMB + SZ_MEMB;
static constexpr size_t SZ_XS     = (size_t)TMEL * Bb * PRE * 2;      // 13.1 MB bf16
static constexpr size_t OFF_PMEM  = OFF_XS + SZ_XS;
static constexpr size_t SZ_PMEM   = (size_t)Bb * TENC * ADIM * 4;     // 6.6 MB f32
static constexpr size_t OFF_BATT  = OFF_PMEM + SZ_PMEM;
static constexpr size_t OFF_BDEC  = OFF_BATT + GATES * 4;
static constexpr size_t OFF_STATE = OFF_BDEC + GATES * 4;
// state block (zero-initialized every launch), f32 unless noted
static constexpr size_t ST_AH   = 0;
static constexpr size_t ST_AC   = ST_AH  + (size_t)Bb*RNN*4;
static constexpr size_t ST_DH   = ST_AC  + (size_t)Bb*RNN*4;
static constexpr size_t ST_DC   = ST_DH  + (size_t)Bb*RNN*4;
static constexpr size_t ST_CTX  = ST_DC  + (size_t)Bb*RNN*4;
static constexpr size_t ST_AW   = ST_CTX + (size_t)Bb*EDIM*4;
static constexpr size_t ST_AWC  = ST_AW  + (size_t)Bb*TENC*4;
static constexpr size_t ST_EBUF = ST_AWC + (size_t)Bb*TENC*4;
static constexpr size_t ST_XATT = ST_EBUF+ (size_t)Bb*TENC*4;         // bf16
static constexpr size_t ST_XDEC = ST_XATT+ (size_t)Bb*KATT*2;         // bf16
static constexpr size_t STATE_BYTES = ST_XDEC + (size_t)Bb*KDEC*2;

// ---------------- params ----------------
struct P {
  const float *mem, *dec_in, *pw1, *pw2, *awih, *awhh, *abih, *abhh;
  const float *qw, *memw, *vw, *lconvw, *densew;
  const float *dwih, *dwhh, *dbih, *dbhh, *projw, *projb, *gatew, *gateb;
  const int   *memlen;
  bf16 *Watt, *Wdec, *memb, *xs, *xcat_att, *xcat_dec;
  float *pmem, *batt, *bdec, *ah, *ac, *dh, *dc, *ctx, *aw, *awc, *ebuf;
  unsigned *cnt, *gen;
  float *out_mel, *out_gate, *out_align;
};

// ---------------- helpers ----------------
__device__ __forceinline__ float sigf(float x) { return 1.f / (1.f + __expf(-x)); }

// Load one lane's 16 bf16 for a 16x32 (16-bit) WMMA operand tile.
// Lane pattern (ISA 7.12.2): K = half*8 + 0..7 then 16 + half*8 + 0..7
__device__ __forceinline__ bfx16 ld_tile(const bf16* row, int kbase, int half) {
  const bfx8* pp = reinterpret_cast<const bfx8*>(row + kbase + half * 8);
  BF16x16 u;
  u.h[0] = pp[0];       // 16 bytes @ kbase + half*8
  u.h[1] = pp[2];       // 16 bytes @ kbase + half*8 + 16
  return u.v;
}

__device__ __forceinline__ fx8 wmma_bf16(bfx16 a, bfx16 b, fx8 c) {
  return __builtin_amdgcn_wmma_f32_16x16x32_bf16(false, a, false, b, (short)0, c,
                                                 false, false);
}

__device__ __forceinline__ void grid_sync(unsigned* cnt, unsigned* gen, unsigned nwg) {
  __syncthreads();
  if (threadIdx.x == 0) {
    volatile unsigned* vgen = (volatile unsigned*)gen;
    unsigned g = *vgen;
    __threadfence();
    if (atomicAdd(cnt, 1u) == nwg - 1u) {
      atomicExch(cnt, 0u);
      __threadfence();
      atomicAdd(gen, 1u);
    } else {
      while (*vgen == g) { __builtin_amdgcn_s_sleep(2); }
    }
    __threadfence();
  }
  __syncthreads();
}

// ---------------- shared memory (32 KB) ----------------
union SMem {
  float red[4][NWAVE][TPB];                                   // 32 KB GEMM reduce
  struct { float convw[LF*2*LK]; float dense[ADIM*LF];
           float v[ADIM];        float pq[ADIM]; } att;       // energies phase
  struct { float e[TENC]; float aw[TENC]; float red[TPB]; } sm; // softmax phase
};

// ---------------- fused LSTM tile phase (WMMA bf16, double-buffered) ----------
// Each WG owns one (m-tile, j-tile); NWAVE waves split K; 4 gate tiles (i,f,g,o)
// accumulate in v8f. Software pipeline: issue the 5 tile loads for k+1, then run
// the 4 WMMAs on the already-resident k tiles -> waits cover in-flight loads
// only (s_wait_loadcnt N instead of 0). LDS tree-reduce; pointwise LSTM fused.
template <int K, int KSPW>
__device__ __forceinline__ void lstm_phase(
    const bf16* Xc, const bf16* W, const float* bias,
    float* hS, float* cS,
    bf16* o1, int s1, int off1, bf16* o2, int s2, int off2,
    float (*red)[NWAVE][TPB])
{
  const int wg = blockIdx.x, tid = threadIdx.x;
  const int wave = tid >> 5, lane = tid & 31;
  const int jt = wg >> 1, m = wg & 1;
  const int rc = lane & 15, half = lane >> 4;

  const bf16* Arow = Xc + (size_t)(m * 16 + rc) * K;
  const bf16* B0 = W + (size_t)(0 * RNN + jt * 16 + rc) * K;
  const bf16* B1 = W + (size_t)(1 * RNN + jt * 16 + rc) * K;
  const bf16* B2 = W + (size_t)(2 * RNN + jt * 16 + rc) * K;
  const bf16* B3 = W + (size_t)(3 * RNN + jt * 16 + rc) * K;

  fx8 z = {0.f,0.f,0.f,0.f,0.f,0.f,0.f,0.f};
  fx8 c0 = z, c1 = z, c2 = z, c3 = z;

  const int kb0 = wave * KSPW * 32;
  // prologue: tiles for first k-step
  bfx16 a_c  = ld_tile(Arow, kb0, half);
  bfx16 b0_c = ld_tile(B0,  kb0, half);
  bfx16 b1_c = ld_tile(B1,  kb0, half);
  bfx16 b2_c = ld_tile(B2,  kb0, half);
  bfx16 b3_c = ld_tile(B3,  kb0, half);
#pragma unroll
  for (int i = 1; i < KSPW; ++i) {
    const int kb = kb0 + i * 32;
    // issue next-step loads first (in flight during the 4 WMMAs below)
    bfx16 a_n  = ld_tile(Arow, kb, half);
    bfx16 b0_n = ld_tile(B0,  kb, half);
    bfx16 b1_n = ld_tile(B1,  kb, half);
    bfx16 b2_n = ld_tile(B2,  kb, half);
    bfx16 b3_n = ld_tile(B3,  kb, half);
    c0 = wmma_bf16(a_c, b0_c, c0);
    c1 = wmma_bf16(a_c, b1_c, c1);
    c2 = wmma_bf16(a_c, b2_c, c2);
    c3 = wmma_bf16(a_c, b3_c, c3);
    a_c = a_n; b0_c = b0_n; b1_c = b1_n; b2_c = b2_n; b3_c = b3_n;
  }
  // epilogue
  c0 = wmma_bf16(a_c, b0_c, c0);
  c1 = wmma_bf16(a_c, b1_c, c1);
  c2 = wmma_bf16(a_c, b2_c, c2);
  c3 = wmma_bf16(a_c, b3_c, c3);

#pragma unroll
  for (int v = 0; v < 8; ++v) {
    red[0][wave][v * 32 + lane] = c0[v];
    red[1][wave][v * 32 + lane] = c1[v];
    red[2][wave][v * 32 + lane] = c2[v];
    red[3][wave][v * 32 + lane] = c3[v];
  }
  __syncthreads();
  float s0 = 0.f, s1v = 0.f, s2v = 0.f, s3v = 0.f;
#pragma unroll
  for (int w = 0; w < NWAVE; ++w) {
    s0  += red[0][w][tid]; s1v += red[1][w][tid];
    s2v += red[2][w][tid]; s3v += red[3][w][tid];
  }
  // decode C/D slot: VGPR v, lane<16 -> row=v ; lane>=16 -> row=v+8
  const int v = tid >> 5, ln = tid & 31;
  const int r = (ln < 16) ? v : v + 8;
  const int col = ln & 15;
  const int bidx = m * 16 + r;
  const int hc = jt * 16 + col;
  const float ig = sigf (s0  + bias[0 * RNN + hc]);
  const float fg = sigf (s1v + bias[1 * RNN + hc]);
  const float gg = tanhf(s2v + bias[2 * RNN + hc]);
  const float og = sigf (s3v + bias[3 * RNN + hc]);
  const size_t si = (size_t)bidx * RNN + hc;
  const float cn = fg * cS[si] + ig * gg;
  const float hn = og * tanhf(cn);
  cS[si] = cn;
  hS[si] = hn;
  o1[(size_t)bidx * s1 + off1 + hc] = (bf16)hn;
  if (o2) o2[(size_t)bidx * s2 + off2 + hc] = (bf16)hn;
}

// ---------------- setup kernels ----------------
__global__ void zero_kernel(unsigned* sync, float* state, size_t nwords) {
  size_t i  = (size_t)blockIdx.x * blockDim.x + threadIdx.x;
  size_t gs = (size_t)gridDim.x * blockDim.x;
  if (i < 2) sync[i] = 0u;
  for (size_t j = i; j < nwords; j += gs) state[j] = 0.f;
}

__global__ void convert_kernel(P p) {
  size_t i0 = (size_t)blockIdx.x * blockDim.x + threadIdx.x;
  size_t gs = (size_t)gridDim.x * blockDim.x;
  for (size_t i = i0; i < (size_t)GATES * KATT; i += gs) {
    int r = (int)(i / KATT), c = (int)(i % KATT);
    float v = (c < RNN) ? p.awih[(size_t)r * 1024 + c] : p.awhh[(size_t)r * 1024 + (c - RNN)];
    p.Watt[i] = (bf16)v;
  }
  for (size_t i = i0; i < (size_t)GATES * KDEC; i += gs) {
    int r = (int)(i / KDEC), c = (int)(i % KDEC);
    float v = (c < RNN + EDIM) ? p.dwih[(size_t)r * 1792 + c] : p.dwhh[(size_t)r * 1024 + (c - 1792)];
    p.Wdec[i] = (bf16)v;
  }
  for (size_t i = i0; i < (size_t)Bb * TENC * EDIM; i += gs) p.memb[i] = (bf16)p.mem[i];
  for (size_t i = i0; i < (size_t)GATES; i += gs) {
    p.batt[i] = p.abih[i] + p.abhh[i];
    p.bdec[i] = p.dbih[i] + p.dbhh[i];
  }
}

__global__ __launch_bounds__(TPB) void prenet_kernel(P p) {
  __shared__ float fr[NMEL];
  __shared__ float l1[PRE];
  const int t = blockIdx.x / Bb, b = blockIdx.x % Bb;
  if (threadIdx.x < NMEL)
    fr[threadIdx.x] = (t == 0) ? 0.f
                               : p.dec_in[((size_t)b * NMEL + threadIdx.x) * TMEL + (t - 1)];
  __syncthreads();
  const int j = threadIdx.x;                  // TPB == PRE
  float s = 0.f;
  for (int k = 0; k < NMEL; ++k) s += fr[k] * p.pw1[(size_t)j * NMEL + k];
  l1[j] = fmaxf(s, 0.f);
  __syncthreads();
  float s2 = 0.f;
  for (int k = 0; k < PRE; ++k) s2 += l1[k] * p.pw2[(size_t)j * PRE + k];
  p.xs[((size_t)t * Bb + b) * PRE + j] = (bf16)fmaxf(s2, 0.f);
}

__global__ void pmem_kernel(P p) {
  const int bt = blockIdx.x;                  // b*TENC + t
  const int a  = threadIdx.x;                 // 128 threads
  const float* mrow = p.mem + (size_t)bt * EDIM;
  const float* wrow = p.memw + (size_t)a * EDIM;
  float s = 0.f;
  for (int k = 0; k < EDIM; ++k) s += mrow[k] * wrow[k];
  p.pmem[(size_t)bt * ADIM + a] = s;
}

// ---------------- persistent decoder kernel ----------------
__global__ __launch_bounds__(TPB) void decoder_kernel(P p) {
  __shared__ SMem sh;
  const int wg = blockIdx.x, tid = threadIdx.x;

  // seed xcat_att with prenet(frame 0); ctx/ah portions already zeroed
  if (wg < Bb) p.xcat_att[(size_t)wg * KATT + tid] = p.xs[(size_t)wg * PRE + tid];
  grid_sync(p.cnt, p.gen, NWG);

  for (int t = 0; t < TMEL; ++t) {
    // ---- P1: attention LSTM (WMMA) -> ah, ac, ah->xcat_att[1024+], xcat_dec[0+]
    lstm_phase<KATT, KS_ATT>(p.xcat_att, p.Watt, p.batt, p.ah, p.ac,
                             p.xcat_att, KATT, PRE + EDIM,
                             p.xcat_dec, KDEC, 0, sh.red);
    grid_sync(p.cnt, p.gen, NWG);

    // ---- P2: energies over (b, 100-position slice)
    {
      const int b = wg & 31, q = wg >> 5;
      for (int i = tid; i < LF * 2 * LK; i += TPB) sh.att.convw[i] = p.lconvw[i];
      for (int i = tid; i < ADIM * LF;   i += TPB) sh.att.dense[i] = p.densew[i];
      for (int i = tid; i < ADIM;        i += TPB) sh.att.v[i]     = p.vw[i];
      if (tid < ADIM) {
        const float* qr = p.qw + (size_t)tid * RNN;
        const float* hr = p.ah + (size_t)b * RNN;
        float s = 0.f;
        for (int k = 0; k < RNN; ++k) s += qr[k] * hr[k];
        sh.att.pq[tid] = s;
      }
      __syncthreads();
      const int len = p.memlen[b];
      for (int tp = tid; tp < 100; tp += TPB) {
        const int tpos = q * 100 + tp;
        float loc[LF];
#pragma unroll
        for (int f = 0; f < LF; ++f) loc[f] = 0.f;
        const float* awr  = p.aw  + (size_t)b * TENC;
        const float* awcr = p.awc + (size_t)b * TENC;
        for (int k = 0; k < LK; ++k) {
          const int idx = tpos + k - LK / 2;
          const bool in = (idx >= 0 && idx < TENC);
          const float cc0 = in ? awr[idx]  : 0.f;
          const float cc1 = in ? awcr[idx] : 0.f;
#pragma unroll
          for (int f = 0; f < LF; ++f)
            loc[f] += cc0 * sh.att.convw[f * (2 * LK) + k]
                    + cc1 * sh.att.convw[f * (2 * LK) + LK + k];
        }
        const float* pm = p.pmem + ((size_t)b * TENC + tpos) * ADIM;
        float e = 0.f;
        for (int a = 0; a < ADIM; ++a) {
          float d = 0.f;
#pragma unroll
          for (int f = 0; f < LF; ++f) d += sh.att.dense[a * LF + f] * loc[f];
          e += sh.att.v[a] * tanhf(sh.att.pq[a] + pm[a] + d);
        }
        if (tpos >= len) e = -1e9f;
        p.ebuf[(size_t)b * TENC + tpos] = e;
      }
    }
    grid_sync(p.cnt, p.gen, NWG);

    // ---- P3: softmax + context + bf16 repack (one WG per batch)
    if (wg < Bb) {
      const int b = wg;
      float mx = -1e30f;
      for (int i = tid; i < TENC; i += TPB) {
        float v = p.ebuf[(size_t)b * TENC + i];
        sh.sm.e[i] = v;
        mx = fmaxf(mx, v);
      }
      sh.sm.red[tid] = mx; __syncthreads();
      for (int s = TPB / 2; s > 0; s >>= 1) {
        if (tid < s) sh.sm.red[tid] = fmaxf(sh.sm.red[tid], sh.sm.red[tid + s]);
        __syncthreads();
      }
      mx = sh.sm.red[0]; __syncthreads();
      float ls = 0.f;
      for (int i = tid; i < TENC; i += TPB) {
        float w = __expf(sh.sm.e[i] - mx);
        sh.sm.aw[i] = w;
        ls += w;
      }
      sh.sm.red[tid] = ls; __syncthreads();
      for (int s = TPB / 2; s > 0; s >>= 1) {
        if (tid < s) sh.sm.red[tid] += sh.sm.red[tid + s];
        __syncthreads();
      }
      const float inv = 1.f / sh.sm.red[0]; __syncthreads();
      for (int i = tid; i < TENC; i += TPB) {
        float w = sh.sm.aw[i] * inv;
        sh.sm.aw[i] = w;
        p.aw[(size_t)b * TENC + i] = w;
        float c = p.awc[(size_t)b * TENC + i] + w;
        p.awc[(size_t)b * TENC + i] = c;
        p.out_align[((size_t)b * TMEL + t) * TENC + i] = w;
      }
      __syncthreads();
      const bf16* mb = p.memb + (size_t)b * TENC * EDIM;
      for (int d = tid; d < EDIM; d += TPB) {
        float acc = 0.f;
        for (int tt = 0; tt < TENC; ++tt)
          acc += sh.sm.aw[tt] * (float)mb[(size_t)tt * EDIM + d];
        p.ctx[(size_t)b * EDIM + d] = acc;
        const bf16 hb = (bf16)acc;
        p.xcat_att[(size_t)b * KATT + PRE + d] = hb;
        p.xcat_dec[(size_t)b * KDEC + RNN + d] = hb;
      }
      if (t + 1 < TMEL)  // prenet frame for next step
        p.xcat_att[(size_t)b * KATT + tid] = p.xs[((size_t)(t + 1) * Bb + b) * PRE + tid];
    }
    grid_sync(p.cnt, p.gen, NWG);

    // ---- P4: decoder LSTM (WMMA) -> dh, dc, dh->xcat_dec[1792+]
    lstm_phase<KDEC, KS_DEC>(p.xcat_dec, p.Wdec, p.bdec, p.dh, p.dc,
                             p.xcat_dec, KDEC, RNN + EDIM,
                             (bf16*)nullptr, 0, 0, sh.red);
    grid_sync(p.cnt, p.gen, NWG);

    // ---- P5: mel projection + gate
    {
      const int gid = wg * TPB + tid;
      if (gid < Bb * NMEL) {
        const int b = gid / NMEL, n = gid % NMEL;
        const float* wrow = p.projw + (size_t)n * (RNN + EDIM);
        const float* hr = p.dh  + (size_t)b * RNN;
        const float* cr = p.ctx + (size_t)b * EDIM;
        float s = p.projb[n];
        for (int k = 0; k < RNN;  ++k) s += hr[k] * wrow[k];
        for (int k = 0; k < EDIM; ++k) s += cr[k] * wrow[RNN + k];
        p.out_mel[((size_t)b * NMEL + n) * TMEL + t] = s;
      } else if (gid < Bb * NMEL + Bb) {
        const int b = gid - Bb * NMEL;
        const float* hr = p.dh  + (size_t)b * RNN;
        const float* cr = p.ctx + (size_t)b * EDIM;
        float s = p.gateb[0];
        for (int k = 0; k < RNN;  ++k) s += hr[k] * p.gatew[k];
        for (int k = 0; k < EDIM; ++k) s += cr[k] * p.gatew[RNN + k];
        p.out_gate[(size_t)b * TMEL + t] = s;
      }
    }
    grid_sync(p.cnt, p.gen, NWG);
  }
}

// ---------------- host launch ----------------
extern "C" void kernel_launch(void* const* d_in, const int* in_sizes, int n_in,
                              void* d_out, int out_size, void* d_ws, size_t ws_size,
                              hipStream_t stream) {
  (void)in_sizes; (void)n_in; (void)out_size; (void)ws_size;
  P p{};
  p.mem    = (const float*)d_in[0];
  p.dec_in = (const float*)d_in[1];
  // d_in[2] = f0s (unused by reference)
  p.pw1    = (const float*)d_in[3];
  p.pw2    = (const float*)d_in[4];
  p.awih   = (const float*)d_in[5];
  p.awhh   = (const float*)d_in[6];
  p.abih   = (const float*)d_in[7];
  p.abhh   = (const float*)d_in[8];
  p.qw     = (const float*)d_in[9];
  p.memw   = (const float*)d_in[10];
  p.vw     = (const float*)d_in[11];
  p.lconvw = (const float*)d_in[12];
  p.densew = (const float*)d_in[13];
  p.dwih   = (const float*)d_in[14];
  p.dwhh   = (const float*)d_in[15];
  p.dbih   = (const float*)d_in[16];
  p.dbhh   = (const float*)d_in[17];
  p.projw  = (const float*)d_in[18];
  p.projb  = (const float*)d_in[19];
  p.gatew  = (const float*)d_in[20];
  p.gateb  = (const float*)d_in[21];
  p.memlen = (const int*)d_in[22];

  char* ws = (char*)d_ws;
  p.cnt  = (unsigned*)ws;
  p.gen  = p.cnt + 1;
  p.Watt = (bf16*)(ws + OFF_WATT);
  p.Wdec = (bf16*)(ws + OFF_WDEC);
  p.memb = (bf16*)(ws + OFF_MEMB);
  p.xs   = (bf16*)(ws + OFF_XS);
  p.pmem = (float*)(ws + OFF_PMEM);
  p.batt = (float*)(ws + OFF_BATT);
  p.bdec = (float*)(ws + OFF_BDEC);
  char* st = ws + OFF_STATE;
  p.ah   = (float*)(st + ST_AH);
  p.ac   = (float*)(st + ST_AC);
  p.dh   = (float*)(st + ST_DH);
  p.dc   = (float*)(st + ST_DC);
  p.ctx  = (float*)(st + ST_CTX);
  p.aw   = (float*)(st + ST_AW);
  p.awc  = (float*)(st + ST_AWC);
  p.ebuf = (float*)(st + ST_EBUF);
  p.xcat_att = (bf16*)(st + ST_XATT);
  p.xcat_dec = (bf16*)(st + ST_XDEC);

  p.out_mel   = (float*)d_out;
  p.out_gate  = p.out_mel  + (size_t)Bb * NMEL * TMEL;
  p.out_align = p.out_gate + (size_t)Bb * TMEL;

  zero_kernel<<<512, TPB, 0, stream>>>(p.cnt, (float*)st, STATE_BYTES / 4);
  convert_kernel<<<8192, TPB, 0, stream>>>(p);
  prenet_kernel<<<TMEL * Bb, TPB, 0, stream>>>(p);
  pmem_kernel<<<Bb * TENC, ADIM, 0, stream>>>(p);
  decoder_kernel<<<NWG, TPB, 0, stream>>>(p);
}
